// GraphLayer_72567767433251
// MI455X (gfx1250) — matrix operand order repaired and checked by reading, not compiled, not run on hardware
//
#include <hip/hip_runtime.h>
#include <stdint.h>

typedef __bf16 bf16;
typedef __attribute__((ext_vector_type(16))) __bf16 v16bf;
typedef __attribute__((ext_vector_type(8)))  __bf16 v8bf;
typedef __attribute__((ext_vector_type(8)))  float  v8f;

constexpr int HH = 2048;          // hidden size
constexpr int CT = HH / 16;       // 128 column tiles per matrix

__device__ __forceinline__ float sigf(float x) { return 1.0f / (1.0f + __expf(-x)); }

// Clamped exp-based tanh: uses hardware v_exp_f32; |err| << bf16 rounding noise.
__device__ __forceinline__ float tanh_fast(float x) {
  const float cx = fminf(fmaxf(x, -15.0f), 15.0f);
  const float e2 = __expf(2.0f * cx);
  return (e2 - 1.0f) / (e2 + 1.0f);
}

__device__ __forceinline__ v8f wmma_bf(v16bf a, v16bf b, v8f c) {
  // D = A(16x32 bf16) * B(32x16 bf16) + C(16x16 f32)
  return __builtin_amdgcn_wmma_f32_16x16x32_bf16(false, a, false, b, (short)0, c, false, false);
}

// A-fragment: lane<16 holds row r, K in {0..15}; lane>=16 K in {16..31}.
// Caller passes row pointer already offset by (half?16:0); we load +0 and +8.
__device__ __forceinline__ v16bf load_a_frag(const bf16* __restrict__ p) {
  v8bf lo = *(const v8bf*)(p);
  v8bf hi = *(const v8bf*)(p + 8);
  v16bf r;
#pragma unroll
  for (int i = 0; i < 8; ++i) { r[i] = lo[i]; r[i + 8] = hi[i]; }
  return r;
}

// ---------------------------------------------------------------------------
// Fused GRU step: out = (1-z)*n + z*h with gi = X@Wih^T + bih, gh = Hm@Whh^T + bhh
// One wave = one 16x16 tile; 6 WMMA chains over K=2048.
// Block mapping: all 8 waves of a block share one column tile (same weight
// rows -> WGP$ hits on the B stream), covering 8 consecutive row tiles.
// Grid: CT x ceil(rowTiles/8) blocks, blockIdx = rtg*CT + jt with jt fastest.
// ---------------------------------------------------------------------------
__global__ __launch_bounds__(256) void gru_tile_kernel(
    const bf16* __restrict__ X,  const int* __restrict__ xidx, int xoff,
    const bf16* __restrict__ Hm, const float* __restrict__ Hf,
    const bf16* __restrict__ Wih, const bf16* __restrict__ Whh,
    const float* __restrict__ bih, const float* __restrict__ bhh,
    float* __restrict__ outF, bf16* __restrict__ outB,
    float* __restrict__ aggr, const int* __restrict__ aidx, int aoff,
    int Brows)
{
  const int lane = threadIdx.x & 31;
  const int wib  = threadIdx.x >> 5;              // wave in block: row-tile select
  const int rowTiles = (Brows + 15) >> 4;
  const int jt = blockIdx.x % CT;                 // shared column tile
  const int rt = (blockIdx.x / CT) * 8 + wib;     // per-wave row tile
  if (rt >= rowTiles) return;

  const int m0   = rt << 4;
  const int half = (lane >> 4) & 1;
  const int lr   = lane & 15;

  // A row for this lane (clamped for guard rows; stores are masked later)
  const int arow  = m0 + lr;
  const int arowc = arow < Brows ? arow : 0;
  const size_t xr = xidx ? (size_t)xidx[2 * arowc + xoff] : (size_t)arowc;
  const bf16* xp = X  + xr * (size_t)HH + (half ? 16 : 0);
  const bf16* hp = Hm + (size_t)arowc * HH + (half ? 16 : 0);

  // B fragments: lane holds W row (gate*H + ncol), contiguous K chunk
  const int ncol = (jt << 4) + lr;
  const int kb   = half ? 16 : 0;
  const bf16* wir = Wih + (size_t)(0 * HH + ncol) * HH + kb;
  const bf16* wiz = Wih + (size_t)(1 * HH + ncol) * HH + kb;
  const bf16* win = Wih + (size_t)(2 * HH + ncol) * HH + kb;
  const bf16* vhr = Whh + (size_t)(0 * HH + ncol) * HH + kb;
  const bf16* vhz = Whh + (size_t)(1 * HH + ncol) * HH + kb;
  const bf16* vhn = Whh + (size_t)(2 * HH + ncol) * HH + kb;

  v8f acc_ir = {0,0,0,0,0,0,0,0};
  v8f acc_iz = acc_ir, acc_in = acc_ir, acc_hr = acc_ir, acc_hz = acc_ir, acc_hn = acc_ir;

#pragma unroll 2
  for (int k0 = 0; k0 < HH; k0 += 32) {
    v16bf ax = load_a_frag(xp + k0);
    v16bf ah = load_a_frag(hp + k0);
    v16bf f_ir = *(const v16bf*)(wir + k0);
    v16bf f_iz = *(const v16bf*)(wiz + k0);
    v16bf f_in = *(const v16bf*)(win + k0);
    v16bf f_hr = *(const v16bf*)(vhr + k0);
    v16bf f_hz = *(const v16bf*)(vhz + k0);
    v16bf f_hn = *(const v16bf*)(vhn + k0);
    acc_ir = wmma_bf(ax, f_ir, acc_ir);
    acc_iz = wmma_bf(ax, f_iz, acc_iz);
    acc_in = wmma_bf(ax, f_in, acc_in);
    acc_hr = wmma_bf(ah, f_hr, acc_hr);
    acc_hz = wmma_bf(ah, f_hz, acc_hz);
    acc_hn = wmma_bf(ah, f_hn, acc_hn);
  }

  const float bir = bih[ncol], biz = bih[HH + ncol], bin_ = bih[2 * HH + ncol];
  const float bhr = bhh[ncol], bhz = bhh[HH + ncol], bhn  = bhh[2 * HH + ncol];

#pragma unroll
  for (int v = 0; v < 8; ++v) {
    const int row = m0 + v + (half ? 8 : 0);   // C/D layout: lanes16-31 hold M=8..15
    if (row < Brows) {
      const float ir = acc_ir[v] + bir, iz = acc_iz[v] + biz, inn = acc_in[v] + bin_;
      const float hr = acc_hr[v] + bhr, hz = acc_hz[v] + bhz, hn  = acc_hn[v] + bhn;
      const float r = sigf(ir + hr);
      const float z = sigf(iz + hz);
      const float n = tanh_fast(inn + r * hn);
      const float hv = Hf ? Hf[(size_t)row * HH + ncol] : (float)Hm[(size_t)row * HH + ncol];
      const float o = (1.0f - z) * n + z * hv;
      if (outF) outF[(size_t)row * HH + ncol] = o;
      if (outB) outB[(size_t)row * HH + ncol] = (bf16)o;
      if (aggr) {
        const int t = aidx[2 * row + aoff];
        atomicAdd(aggr + (size_t)t * HH + ncol, o);
      }
    }
  }
}

// ---------------------------------------------------------------------------
// rho partial: s[col] += sum_rows sigmoid(F@Wa^T + ba) * tanh(F@Wb^T + bb)
// Same block mapping as gru_tile_kernel (waves share the weight stream).
// ---------------------------------------------------------------------------
__global__ __launch_bounds__(256) void rho_tile_kernel(
    const bf16* __restrict__ F, const bf16* __restrict__ Wa, const bf16* __restrict__ Wb,
    const float* __restrict__ ba, const float* __restrict__ bb,
    float* __restrict__ s, int Brows)
{
  const int lane = threadIdx.x & 31;
  const int wib  = threadIdx.x >> 5;
  const int rowTiles = (Brows + 15) >> 4;
  const int jt = blockIdx.x % CT;
  const int rt = (blockIdx.x / CT) * 8 + wib;
  if (rt >= rowTiles) return;

  const int m0   = rt << 4;
  const int half = (lane >> 4) & 1;
  const int lr   = lane & 15;

  const int arow  = m0 + lr;
  const int arowc = arow < Brows ? arow : 0;
  const bf16* fp = F + (size_t)arowc * HH + (half ? 16 : 0);

  const int ncol = (jt << 4) + lr;
  const int kb   = half ? 16 : 0;
  const bf16* wa = Wa + (size_t)ncol * HH + kb;
  const bf16* wb = Wb + (size_t)ncol * HH + kb;

  v8f acc_a = {0,0,0,0,0,0,0,0};
  v8f acc_b = acc_a;

#pragma unroll 2
  for (int k0 = 0; k0 < HH; k0 += 32) {
    v16bf af  = load_a_frag(fp + k0);
    v16bf fba = *(const v16bf*)(wa + k0);
    v16bf fbb = *(const v16bf*)(wb + k0);
    acc_a = wmma_bf(af, fba, acc_a);
    acc_b = wmma_bf(af, fbb, acc_b);
  }

  const float bav = ba[ncol], bbv = bb[ncol];
  float partial = 0.0f;
#pragma unroll
  for (int v = 0; v < 8; ++v) {
    const int row = m0 + v + (half ? 8 : 0);
    if (row < Brows)
      partial += sigf(acc_a[v] + bav) * tanh_fast(acc_b[v] + bbv);
  }
  atomicAdd(s + ncol, partial);
}

// ---------------------------------------------------------------------------
// Small helper kernels
// ---------------------------------------------------------------------------
__global__ void cast_f2bf_kernel(const float* __restrict__ s, bf16* __restrict__ d, long long n) {
  long long i = (long long)blockIdx.x * blockDim.x + threadIdx.x;
  const long long stride = (long long)gridDim.x * blockDim.x;
  for (; i < n; i += stride) d[i] = (bf16)s[i];
}

__global__ void fill0_f32_kernel(float* __restrict__ p, long long n) {
  long long i = (long long)blockIdx.x * blockDim.x + threadIdx.x;
  const long long stride = (long long)gridDim.x * blockDim.x;
  for (; i < n; i += stride) p[i] = 0.0f;
}

__global__ void tanh_vec_kernel(const float* __restrict__ s, float* __restrict__ d, int n) {
  const int i = blockIdx.x * blockDim.x + threadIdx.x;
  if (i < n) d[i] = tanh_fast(s[i]);
}

// B=1 GRU step in f32 (tiny: 25M MACs); one thread per output column.
__global__ __launch_bounds__(256) void gru_b1_kernel(
    const float* __restrict__ x, const float* __restrict__ hv,
    const float* __restrict__ Wih, const float* __restrict__ Whh,
    const float* __restrict__ bih, const float* __restrict__ bhh,
    float* __restrict__ out)
{
  const int c = blockIdx.x * blockDim.x + threadIdx.x;
  if (c >= HH) return;
  float ir = 0.f, iz = 0.f, inn = 0.f, hr = 0.f, hz = 0.f, hn = 0.f;
  const float* wr = Wih + (size_t)(0 * HH + c) * HH;
  const float* wz = Wih + (size_t)(1 * HH + c) * HH;
  const float* wn = Wih + (size_t)(2 * HH + c) * HH;
  const float* vr = Whh + (size_t)(0 * HH + c) * HH;
  const float* vz = Whh + (size_t)(1 * HH + c) * HH;
  const float* vn = Whh + (size_t)(2 * HH + c) * HH;
  for (int k = 0; k < HH; ++k) {
    const float xv = x[k], hk = hv[k];
    ir += xv * wr[k]; iz += xv * wz[k]; inn += xv * wn[k];
    hr += hk * vr[k]; hz += hk * vz[k]; hn  += hk * vn[k];
  }
  ir += bih[c]; iz += bih[HH + c]; inn += bih[2 * HH + c];
  hr += bhh[c]; hz += bhh[HH + c]; hn  += bhh[2 * HH + c];
  const float r = sigf(ir + hr);
  const float z = sigf(iz + hz);
  const float n = tanh_fast(inn + r * hn);
  out[c] = (1.f - z) * n + z * hv[c];
}

// ---------------------------------------------------------------------------
// Host orchestration
// ---------------------------------------------------------------------------
static inline void launch_cast(const float* s, bf16* d, long long n, hipStream_t st) {
  cast_f2bf_kernel<<<2048, 256, 0, st>>>(s, d, n);
}

extern "C" void kernel_launch(void* const* d_in, const int* in_sizes, int n_in,
                              void* d_out, int out_size, void* d_ws, size_t ws_size,
                              hipStream_t stream)
{
  (void)n_in; (void)out_size; (void)ws_size;
  const float* node  = (const float*)d_in[0];
  const int*   ee    = (const int*)d_in[1];       // [E,2] (src,dst)
  const float* edge  = (const float*)d_in[2];
  const float* gfeat = (const float*)d_in[3];
  const float* We_ih = (const float*)d_in[4];
  const float* We_hh = (const float*)d_in[5];
  const float* be_ih = (const float*)d_in[6];
  const float* be_hh = (const float*)d_in[7];
  const float* Wv_ih = (const float*)d_in[8];
  const float* Wv_hh = (const float*)d_in[9];
  const float* bv_ih = (const float*)d_in[10];
  const float* bv_hh = (const float*)d_in[11];
  const float* Wu_ih = (const float*)d_in[12];
  const float* Wu_hh = (const float*)d_in[13];
  const float* bu_ih = (const float*)d_in[14];
  const float* bu_hh = (const float*)d_in[15];
  const float* W1 = (const float*)d_in[16]; const float* b1 = (const float*)d_in[17];
  const float* W2 = (const float*)d_in[18]; const float* b2 = (const float*)d_in[19];
  const float* W3 = (const float*)d_in[20]; const float* b3 = (const float*)d_in[21];
  const float* W4 = (const float*)d_in[22]; const float* b4 = (const float*)d_in[23];

  const int N = in_sizes[0] / HH;
  const int E = in_sizes[2] / HH;

  // Output layout: v_new [N*H] | e_new [E*H] | g_new [H]
  float* v_out = (float*)d_out;
  float* e_out = v_out + (size_t)N * HH;
  float* g_out = e_out + (size_t)E * HH;

  // Workspace carve (256B aligned)
  size_t off = 0;
  auto carve = [&](size_t bytes) -> char* {
    char* p = (char*)d_ws + off;
    off += (bytes + 255) & ~(size_t)255;
    return p;
  };
  const size_t w3sz = (size_t)3 * HH * HH;  // 3H*H elements
  const size_t w1sz = (size_t)HH * HH;
  bf16* We_ih_b = (bf16*)carve(w3sz * 2);
  bf16* We_hh_b = (bf16*)carve(w3sz * 2);
  bf16* Wv_ih_b = (bf16*)carve(w3sz * 2);
  bf16* Wv_hh_b = (bf16*)carve(w3sz * 2);
  bf16* W1_b = (bf16*)carve(w1sz * 2);
  bf16* W2_b = (bf16*)carve(w1sz * 2);
  bf16* W3_b = (bf16*)carve(w1sz * 2);
  bf16* W4_b = (bf16*)carve(w1sz * 2);
  bf16* node_b = (bf16*)carve((size_t)N * HH * 2);
  bf16* edge_b = (bf16*)carve((size_t)E * HH * 2);
  bf16* h1_b   = (bf16*)carve((size_t)E * HH * 2);
  bf16* enew_b = (bf16*)carve((size_t)E * HH * 2);
  bf16* aggr_b = (bf16*)carve((size_t)N * HH * 2);
  bf16* vnew_b = (bf16*)carve((size_t)N * HH * 2);
  float* aggr_f = (float*)carve((size_t)N * HH * 4);
  float* s_e = (float*)carve(HH * 4);
  float* s_v = (float*)carve(HH * 4);
  float* ae  = (float*)carve(HH * 4);
  float* av  = (float*)carve(HH * 4);
  float* g1  = (float*)carve(HH * 4);

  // 1) BF16 staging of weights + activations
  launch_cast(We_ih, We_ih_b, (long long)w3sz, stream);
  launch_cast(We_hh, We_hh_b, (long long)w3sz, stream);
  launch_cast(Wv_ih, Wv_ih_b, (long long)w3sz, stream);
  launch_cast(Wv_hh, Wv_hh_b, (long long)w3sz, stream);
  launch_cast(W1, W1_b, (long long)w1sz, stream);
  launch_cast(W2, W2_b, (long long)w1sz, stream);
  launch_cast(W3, W3_b, (long long)w1sz, stream);
  launch_cast(W4, W4_b, (long long)w1sz, stream);
  launch_cast(node, node_b, (long long)N * HH, stream);
  launch_cast(edge, edge_b, (long long)E * HH, stream);

  // 2) Zero accumulators (aggr, rho sums)
  fill0_f32_kernel<<<2048, 256, 0, stream>>>(aggr_f, (long long)N * HH);
  fill0_f32_kernel<<<1, 256, 0, stream>>>(s_e, HH);
  fill0_f32_kernel<<<1, 256, 0, stream>>>(s_v, HH);

  // Grid: jt fastest (so one activation band is consumed by 128 consecutive
  // blocks while the bf16 weights stay L2-resident), 8 row tiles per block.
  const int blkE = CT * (((E + 15) >> 4) + 7) / 8;
  const int blkN = CT * (((N + 15) >> 4) + 7) / 8;

  // 3) h = gru(node[src], edge, We)            -> h1 (bf16 only)
  gru_tile_kernel<<<blkE, 256, 0, stream>>>(
      node_b, ee, /*xoff=*/0, edge_b, edge,
      We_ih_b, We_hh_b, be_ih, be_hh,
      nullptr, h1_b, nullptr, nullptr, 0, E);

  // 4) e_new = gru(node[dst], h, We)           -> e_out, enew_b, fused segsum into aggr_f (by src)
  gru_tile_kernel<<<blkE, 256, 0, stream>>>(
      node_b, ee, /*xoff=*/1, h1_b, nullptr,
      We_ih_b, We_hh_b, be_ih, be_hh,
      e_out, enew_b, aggr_f, ee, /*aoff=*/0, E);

  // 5) aggr -> bf16
  launch_cast(aggr_f, aggr_b, (long long)N * HH, stream);

  // 6) v_new = gru(aggr, node, Wv)             -> v_out, vnew_b
  gru_tile_kernel<<<blkN, 256, 0, stream>>>(
      aggr_b, nullptr, 0, node_b, node,
      Wv_ih_b, Wv_hh_b, bv_ih, bv_hh,
      v_out, vnew_b, nullptr, nullptr, 0, N);

  // 7) ae = tanh(sum sigmoid(e@W3^T+b3)*tanh(e@W4^T+b4)) ; av analog with v_new, W1/W2
  rho_tile_kernel<<<blkE, 256, 0, stream>>>(enew_b, W3_b, W4_b, b3, b4, s_e, E);
  rho_tile_kernel<<<blkN, 256, 0, stream>>>(vnew_b, W1_b, W2_b, b1, b2, s_v, N);
  tanh_vec_kernel<<<(HH + 255) / 256, 256, 0, stream>>>(s_e, ae, HH);
  tanh_vec_kernel<<<(HH + 255) / 256, 256, 0, stream>>>(s_v, av, HH);

  // 8) g = gru(ae, global); g = gru(av, g)     (B=1, f32 scalar path)
  gru_b1_kernel<<<HH / 256, 256, 0, stream>>>(ae, gfeat, Wu_ih, Wu_hh, bu_ih, bu_hh, g1);
  gru_b1_kernel<<<HH / 256, 256, 0, stream>>>(av, g1,    Wu_ih, Wu_hh, bu_ih, bu_hh, g_out);
}
